// SRL_Framenet_2345052144110
// MI455X (gfx1250) — matrix operand order repaired and checked by reading, not compiled, not run on hardware
//
#include <hip/hip_runtime.h>
#include <hip/hip_bf16.h>
#include <math.h>

typedef __attribute__((ext_vector_type(16))) _Float16 v16h;
typedef __attribute__((ext_vector_type(8)))  _Float16 v8h;
typedef __attribute__((ext_vector_type(2)))  _Float16 h2;
typedef __attribute__((ext_vector_type(8)))  float    v8f;

#define NB   16      // batch
#define NT   64      // tokens
#define NC   32      // constituents
#define NN   96      // N = T + C
#define NH   512     // hidden
#define NL_  64      // label count L
#define NEGV (-1e13f)

__device__ __forceinline__ v8f wmma16(v16h a, v16h b, v8f c) {
  return __builtin_amdgcn_wmma_f32_16x16x32_f16(false, a, false, b, (short)0, c,
                                                false, false);
}
__device__ __forceinline__ float sigmf(float x) { return 1.f / (1.f + __expf(-x)); }

// ---- CDNA5 async global->LDS copy (ASYNCcnt path), with safe fallback -------
#if defined(__gfx1250__) &&                                                     \
    __has_builtin(__builtin_amdgcn_global_load_async_to_lds_b128) &&            \
    __has_builtin(__builtin_amdgcn_s_wait_asynccnt)
typedef int v4i_t __attribute__((vector_size(4 * sizeof(int))));
typedef __attribute__((address_space(1))) v4i_t* as1v4p;
typedef __attribute__((address_space(3))) v4i_t* as3v4p;
__device__ __forceinline__ void async_b128(const void* g, void* l) {
  __builtin_amdgcn_global_load_async_to_lds_b128((as1v4p)g, (as3v4p)l, 0, 0);
}
#define WAIT_ASYNC() __builtin_amdgcn_s_wait_asynccnt(0)
#else
__device__ __forceinline__ void async_b128(const void* g, void* l) {
  *(v8h*)l = *(const v8h*)g;
}
#define WAIT_ASYNC()
#endif

// Build a 16x32 f16 A-fragment from LDS (row-major, 'stride' halves per row),
// contiguous K order matching B fragments (lanes 0-15: K 0..15; 16-31: K 16..31).
__device__ __forceinline__ v16h a_frag_lds(const _Float16* base, int stride, int k0,
                                           int lane) {
  v16h a;
  int row = lane & 15;
  int kof = (lane < 16) ? 0 : 16;
#pragma unroll
  for (int j = 0; j < 8; ++j) {
    int kk = k0 + kof + 2 * j;
    a[2 * j]     = base[row * stride + kk];
    a[2 * j + 1] = base[row * stride + kk + 1];
  }
  return a;
}

// ---------------------------------------------------------------------------
// LayerNorm(masked fixed_embs) + indicator embedding concat -> f16 (1024 x 320)
// ---------------------------------------------------------------------------
__global__ __launch_bounds__(128) void k_ln_embed(
    const float* __restrict__ fixed, const int* __restrict__ flags,
    const float* __restrict__ smask, const float* __restrict__ ln_g,
    const float* __restrict__ ln_b, const float* __restrict__ ind_emb,
    _Float16* __restrict__ emb) {
  __shared__ float red[128];
  int bt = blockIdx.x, tid = threadIdx.x;
  float m = smask[bt];
  float x = (tid < 100) ? fixed[bt * 100 + tid] * m : 0.f;
  red[tid] = x;
  __syncthreads();
  for (int o = 64; o; o >>= 1) { if (tid < o) red[tid] += red[tid + o]; __syncthreads(); }
  float mu = red[0] / 100.f;
  __syncthreads();
  float d = (tid < 100) ? (x - mu) : 0.f;
  red[tid] = d * d;
  __syncthreads();
  for (int o = 64; o; o >>= 1) { if (tid < o) red[tid] += red[tid + o]; __syncthreads(); }
  float var = red[0] / 100.f;
  int fl = flags[bt];
  _Float16* er = emb + (size_t)bt * 320;
  if (tid < 100)
    er[tid] = (_Float16)(d * rsqrtf(var + 1e-5f) * ln_g[tid] + ln_b[tid]);
  for (int j = tid; j < 200; j += 128) er[100 + j] = (_Float16)ind_emb[fl * 200 + j];
  for (int j = tid; j < 20; j += 128) er[300 + j] = (_Float16)0.f;
}

// ---------------------------------------------------------------------------
// Tiled f32 -> f16 transpose: dst[z][n][k] = src[z][k][n]   (K, Nc mult of 32)
// ---------------------------------------------------------------------------
__global__ __launch_bounds__(256) void k_transpose_f16(
    const float* __restrict__ src, _Float16* __restrict__ dst, int K, int Nc) {
  __shared__ _Float16 t[32][33];
  size_t base = (size_t)blockIdx.z * K * Nc;
  int k0 = blockIdx.y * 32, n0 = blockIdx.x * 32;
  int lx = threadIdx.x & 31, ly = threadIdx.x >> 5;
#pragma unroll
  for (int r = ly; r < 32; r += 8)
    t[r][lx] = (_Float16)src[base + (size_t)(k0 + r) * Nc + n0 + lx];
  __syncthreads();
#pragma unroll
  for (int r = ly; r < 32; r += 8)
    dst[base + (size_t)(n0 + r) * K + k0 + lx] = t[lx][r];
}

// ---------------------------------------------------------------------------
// WMMA GEMM: C(f32,MxN) = act(A(f16,MxK) @ B(f32,KbxN) + bias)
// Block = 4 waves; tile 64(M) x 64(N); double-buffered LDS; A staged via
// async global->LDS b128; B converted f32->f16 into transposed LDS tile
// ([n][k], 40-half stride => aligned, conflict-free v16h fragments).
// 4 WMMA per wave per K-step, one barrier per K-step.
// ---------------------------------------------------------------------------
__device__ __forceinline__ void gemm_stage(
    const _Float16* __restrict__ A, const float* __restrict__ Bm, _Float16* dA,
    _Float16* dB, int m0, int n0, int N, int K, int Kb, int k0, int tid) {
  int arow = tid >> 2, akq = (tid & 3) * 8;  // A: 2x b128 per thread (64x32 tile)
  async_b128(A + (size_t)(m0 + arow) * K + k0 + akq, dA + arow * 32 + akq);
  async_b128(A + (size_t)(m0 + arow + 32) * K + k0 + akq,
             dA + (arow + 32) * 32 + akq);
  int bkk = (tid >> 3) * 2, bnb = (tid & 7) * 8;  // B: 2 k-rows x 8 n per thread
  int gk = k0 + bkk;
  const float* b0 = Bm + (size_t)gk * N + n0 + bnb;
  float r0[8], r1[8];
  if (gk < Kb) {
#pragma unroll
    for (int j = 0; j < 8; ++j) r0[j] = b0[j];
  } else {
#pragma unroll
    for (int j = 0; j < 8; ++j) r0[j] = 0.f;
  }
  if (gk + 1 < Kb) {
#pragma unroll
    for (int j = 0; j < 8; ++j) r1[j] = b0[N + j];
  } else {
#pragma unroll
    for (int j = 0; j < 8; ++j) r1[j] = 0.f;
  }
#pragma unroll
  for (int j = 0; j < 8; ++j) {
    h2 p;
    p.x = (_Float16)r0[j];
    p.y = (_Float16)r1[j];
    *(h2*)&dB[(bnb + j) * 40 + bkk] = p;
  }
}

__global__ __launch_bounds__(128) void k_gemm_f16(
    const _Float16* __restrict__ A, const float* __restrict__ Bm,
    const float* __restrict__ bias, float* __restrict__ C,
    _Float16* __restrict__ Ch, int M, int N, int K, int Kb, int act) {
  __shared__ _Float16 sA[2][64 * 32];
  __shared__ _Float16 sBT[2][64 * 40];
  int tid = threadIdx.x, wv = tid >> 5, l = tid & 31;
  int m0 = blockIdx.y * 64, n0 = blockIdx.x * 64;
  v8f acc[4];
#pragma unroll
  for (int s = 0; s < 4; ++s)
#pragma unroll
    for (int q = 0; q < 8; ++q) acc[s][q] = 0.f;

  gemm_stage(A, Bm, sA[0], sBT[0], m0, n0, N, K, Kb, 0, tid);
  int buf = 0;
  for (int k0 = 0; k0 < K; k0 += 32) {
    WAIT_ASYNC();
    __syncthreads();  // publish staged buf; prior reads of buf^1 done
    if (k0 + 32 < K)
      gemm_stage(A, Bm, sA[buf ^ 1], sBT[buf ^ 1], m0, n0, N, K, Kb, k0 + 32, tid);
    v16h b = *(const v16h*)&sBT[buf][(wv * 16 + (l & 15)) * 40 + ((l < 16) ? 0 : 16)];
#pragma unroll
    for (int s = 0; s < 4; ++s) {
      v16h a = a_frag_lds(&sA[buf][s * 16 * 32], 32, 0, l);
      acc[s] = wmma16(a, b, acc[s]);
    }
    buf ^= 1;
  }
  int colc = n0 + wv * 16 + (l & 15);
#pragma unroll
  for (int s = 0; s < 4; ++s) {
#pragma unroll
    for (int r = 0; r < 8; ++r) {
      int m = m0 + s * 16 + r + 8 * (l >> 4);
      float v = acc[s][r];
      if (bias) v += bias[colc];
      if (act) v = fmaxf(v, 0.f);
      C[(size_t)m * N + colc] = v;
      if (Ch) Ch[(size_t)m * N + colc] = (_Float16)v;
    }
  }
}

// ---------------------------------------------------------------------------
// Highway-LSTM recurrent scan: one persistent workgroup (16 waves).
// h in LDS (f16, WMMA A operand); cell state c in VGPRs; Ws pre-transposed to
// f16 [n][k] so each B-fragment is one contiguous v16h (2x global_load_b128).
// ---------------------------------------------------------------------------
__global__ __launch_bounds__(512) void k_lstm_scan(
    const _Float16* __restrict__ WsT,  // (2560 x 512) f16, [n][k]
    const float* __restrict__ bias,    // (2560)
    const float* __restrict__ px,      // (B,T,3072) = [i f g o hw x]
    float* __restrict__ houtf,         // (B,T,512)
    _Float16* __restrict__ houth, int reverse) {
  __shared__ _Float16 hsh[16 * NH];
  int tid = threadIdx.x, wv = tid >> 5, l = tid & 31;
  for (int i = tid; i < 16 * NH; i += 512) hsh[i] = (_Float16)0.f;
  int hi = l >> 4;
  int col[2] = {wv * 32 + (l & 15), wv * 32 + 16 + (l & 15)};
  float c0[2][8];
#pragma unroll
  for (int t = 0; t < 2; ++t)
#pragma unroll
    for (int r = 0; r < 8; ++r) c0[t][r] = 0.f;
  float breg[5][2];
#pragma unroll
  for (int g = 0; g < 5; ++g)
#pragma unroll
    for (int t = 0; t < 2; ++t) breg[g][t] = bias[g * NH + col[t]];
  __syncthreads();

  for (int s = 0; s < NT; ++s) {
    int tt = reverse ? (NT - 1 - s) : s;
    v8f acc[5][2];
#pragma unroll
    for (int g = 0; g < 5; ++g)
#pragma unroll
      for (int t = 0; t < 2; ++t)
#pragma unroll
        for (int q = 0; q < 8; ++q) acc[g][t][q] = 0.f;

    for (int k0 = 0; k0 < NH; k0 += 32) {
      v16h a = a_frag_lds(hsh, NH, k0, l);
      int kb = k0 + ((l < 16) ? 0 : 16);
#pragma unroll
      for (int g = 0; g < 5; ++g)
#pragma unroll
        for (int t = 0; t < 2; ++t) {
          v16h b = *(const v16h*)(WsT + (size_t)(g * NH + col[t]) * NH + kb);
          acc[g][t] = wmma16(a, b, acc[g][t]);
        }
    }
    __syncthreads();  // all waves done reading h
#pragma unroll
    for (int t = 0; t < 2; ++t) {
      int cc = col[t];
#pragma unroll
      for (int r = 0; r < 8; ++r) {
        int bi = r + 8 * hi;
        size_t pxb = ((size_t)bi * NT + tt) * 3072;
        float zi = acc[0][t][r] + px[pxb + 0 * NH + cc] + breg[0][t];
        float zf = acc[1][t][r] + px[pxb + 1 * NH + cc] + breg[1][t];
        float zg = acc[2][t][r] + px[pxb + 2 * NH + cc] + breg[2][t];
        float zo = acc[3][t][r] + px[pxb + 3 * NH + cc] + breg[3][t];
        float zh = acc[4][t][r] + px[pxb + 4 * NH + cc] + breg[4][t];
        float hx = px[pxb + 5 * NH + cc];
        float cnew = sigmf(zf) * c0[t][r] + sigmf(zi) * tanhf(zg);
        c0[t][r] = cnew;
        float hw = sigmf(zh);
        float hnew = hw * (sigmf(zo) * tanhf(cnew)) + (1.f - hw) * hx;
        hsh[bi * NH + cc] = (_Float16)hnew;
        size_t oi = ((size_t)bi * NT + tt) * NH + cc;
        houtf[oi] = hnew;
        houth[oi] = (_Float16)hnew;
      }
    }
    __syncthreads();  // publish h for next step
  }
}

// gate matvec: out[r] = dot(X[r,:512], v)
__global__ __launch_bounds__(32) void k_matvec512(const float* __restrict__ X,
                                                  const float* __restrict__ v,
                                                  float* __restrict__ out) {
  int r = blockIdx.x, l = threadIdx.x;
  float s = 0.f;
  for (int k = l; k < NH; k += 32) s += X[(size_t)r * NH + k] * v[k];
  for (int o = 16; o; o >>= 1) s += __shfl_down(s, o, 32);
  if (l == 0) out[r] = s;
}

// gcn_in = concat(lstm_out, constituents) -> f32 + f16
__global__ void k_build_flat(const float* __restrict__ hseq,
                             const float* __restrict__ consts,
                             float* __restrict__ gf, _Float16* __restrict__ gh) {
  int i = blockIdx.x * 256 + threadIdx.x;
  if (i >= NB * NN * NH) return;
  int row = i >> 9, h = i & 511;
  int b = row / NN, n = row % NN;
  float v = (n < NT) ? hseq[((size_t)b * NT + n) * NH + h]
                     : consts[((size_t)b * NC + (n - NT)) * NH + h];
  gf[i] = v;
  gh[i] = (_Float16)v;
}

// GCN gather/gate/combine (+relu, node mask, residual)
__global__ __launch_bounds__(128) void k_gcn_combine(
    const float* __restrict__ xf, const float* __restrict__ pin,
    const float* __restrict__ pout, const float* __restrict__ ploop,
    const float* __restrict__ bin, const float* __restrict__ bout,
    const float* __restrict__ gin, const float* __restrict__ gout,
    const float* __restrict__ gloop, const float* __restrict__ bing,
    const float* __restrict__ boutg, const int* __restrict__ arcin,
    const int* __restrict__ arcout, const int* __restrict__ labin,
    const int* __restrict__ labout, const float* __restrict__ min_,
    const float* __restrict__ mout, const float* __restrict__ smask,
    const float* __restrict__ cmask, float* __restrict__ of,
    _Float16* __restrict__ oh, int has_loop) {
  int i = blockIdx.x, tid = threadIdx.x;
  int b = i / NN, n = i % NN;
  int iin = arcin[i] * NN + arcin[NB * NN + i];
  int iou = arcout[i] * NN + arcout[NB * NN + i];
  int li = labin[i], lo = labout[i];
  float gi = sigmf(gin[iin] + bing[li]) * min_[i];
  float go = sigmf(gout[iou] + boutg[lo]) * mout[i];
  float gl = has_loop ? sigmf(gloop[i]) : 0.f;
  float nm = (n < NT) ? smask[b * NT + n] : cmask[b * NC + (n - NT)];
  for (int h = tid; h < NH; h += 128) {
    float s = gi * (pin[(size_t)iin * NH + h] + bin[(size_t)li * NH + h]) +
              go * (pout[(size_t)iou * NH + h] + bout[(size_t)lo * NH + h]);
    if (has_loop) s += gl * ploop[(size_t)i * NH + h];
    float r = fmaxf(s, 0.f) * nm + xf[(size_t)i * NH + h];
    of[(size_t)i * NH + h] = r;
    oh[(size_t)i * NH + h] = (_Float16)r;
  }
}

__global__ void k_gather_topin(const _Float16* __restrict__ g,
                               _Float16* __restrict__ dst) {
  int i = blockIdx.x * 256 + threadIdx.x;
  if (i >= NB * NT * NH) return;
  int bt = i >> 9, h = i & 511, b = bt >> 6, t = bt & 63;
  dst[i] = g[(((size_t)b * NN + t) << 9) + h];
}

__global__ void k_gather_preds(const _Float16* __restrict__ h,
                               const int* __restrict__ pidx,
                               _Float16* __restrict__ dst) {
  int i = blockIdx.x * 256 + threadIdx.x;
  if (i >= NB * NT * NH) return;
  int bt = i >> 9;
  dst[i] = h[((size_t)pidx[bt] << 9) + (i & 511)];
}

// scores base: pred.wp[l] + arg.wa[l] + b_bil[l]
__global__ __launch_bounds__(64) void k_bil_linear(
    const float* __restrict__ pred, const float* __restrict__ arg,
    const float* __restrict__ wp, const float* __restrict__ wa,
    const float* __restrict__ bb, float* __restrict__ out) {
  int bt = blockIdx.x, l = threadIdx.x;
  float s = bb[l];
  const float* p = pred + (size_t)bt * NH;
  const float* a = arg + (size_t)bt * NH;
  const float* w1 = wp + (size_t)l * NH;
  const float* w2 = wa + (size_t)l * NH;
  for (int k = 0; k < NH; ++k) s += p[k] * w1[k] + a[k] * w2[k];
  out[(size_t)bt * NL_ + l] = s;
}

// scores += (pred @ U[l]) . arg ; apply constraint mask.
// UT is U pre-transposed to f16 [l][n][k] -> contiguous v16h B-fragments.
__global__ __launch_bounds__(128) void k_bilinear(
    const _Float16* __restrict__ Ph, const _Float16* __restrict__ UT,
    const float* __restrict__ Af, const int* __restrict__ cons,
    float* __restrict__ out) {
  __shared__ _Float16 sA[2][16 * 32];
  __shared__ float part[16];
  int tid = threadIdx.x, wv = tid >> 5, l = tid & 31;
  int m0 = blockIdx.x * 16, li = blockIdx.y;
  if (tid < 16) part[tid] = 0.f;
  v8f acc[8];
#pragma unroll
  for (int nt = 0; nt < 8; ++nt)
#pragma unroll
    for (int q = 0; q < 8; ++q) acc[nt][q] = 0.f;

  if (tid < 64) {
    int r = tid >> 2, kq = (tid & 3) * 8;
    async_b128(Ph + (size_t)(m0 + r) * NH + kq, &sA[0][r * 32 + kq]);
  }
  int buf = 0;
  for (int k0 = 0; k0 < NH; k0 += 32) {
    WAIT_ASYNC();
    __syncthreads();
    if (k0 + 32 < NH && tid < 64) {
      int r = tid >> 2, kq = (tid & 3) * 8;
      async_b128(Ph + (size_t)(m0 + r) * NH + k0 + 32 + kq,
                 &sA[buf ^ 1][r * 32 + kq]);
    }
    v16h a = a_frag_lds(sA[buf], 32, 0, l);
    int kb = k0 + ((l < 16) ? 0 : 16);
#pragma unroll
    for (int nt = 0; nt < 8; ++nt) {
      int colc = wv * 128 + nt * 16 + (l & 15);
      v16h b = *(const v16h*)(UT + ((size_t)li * NH + colc) * NH + kb);
      acc[nt] = wmma16(a, b, acc[nt]);
    }
    buf ^= 1;
  }
  __syncthreads();
#pragma unroll
  for (int r = 0; r < 8; ++r) {
    int row = r + 8 * (l >> 4);
    int m = m0 + row;
    float v = 0.f;
#pragma unroll
    for (int nt = 0; nt < 8; ++nt)
      v += acc[nt][r] * Af[(size_t)m * NH + wv * 128 + nt * 16 + (l & 15)];
    atomicAdd(&part[row], v);  // ds_add_f32
  }
  __syncthreads();
  if (tid < 16) {
    int bt = m0 + tid, b = bt >> 6;
    float lin = out[(size_t)bt * NL_ + li];
    out[(size_t)bt * NL_ + li] =
        (cons[b * NL_ + li] == 0) ? NEGV : (lin + part[tid]);
  }
}

// ---------------------------------------------------------------------------
extern "C" void kernel_launch(void* const* d_in, const int* in_sizes, int n_in,
                              void* d_out, int out_size, void* d_ws, size_t ws_size,
                              hipStream_t stream) {
  (void)in_sizes; (void)n_in; (void)out_size; (void)ws_size;
  auto F = [&](int i) { return (const float*)d_in[i]; };
  auto I = [&](int i) { return (const int*)d_in[i]; };

  char* w = (char*)d_ws;
  auto alloc = [&](size_t bytes) {
    void* p = (void*)w;
    w += (bytes + 255) & ~(size_t)255;
    return p;
  };
  _Float16* embH   = (_Float16*)alloc((size_t)1024 * 320 * 2);
  float*    px     = (float*)alloc((size_t)1024 * 3072 * 4);
  float*    hseqF  = (float*)alloc((size_t)1024 * NH * 4);
  _Float16* hseqH  = (_Float16*)alloc((size_t)1024 * NH * 2);
  _Float16* WsT    = (_Float16*)alloc((size_t)2560 * NH * 2);
  float*    gF0    = (float*)alloc((size_t)1536 * NH * 4);
  float*    gF1    = (float*)alloc((size_t)1536 * NH * 4);
  _Float16* gH     = (_Float16*)alloc((size_t)1536 * NH * 2);
  float*    potIn  = (float*)alloc((size_t)1536 * NH * 4);
  float*    potOut = (float*)alloc((size_t)1536 * NH * 4);
  float*    potLp  = (float*)alloc((size_t)1536 * NH * 4);
  float*    gateIn = (float*)alloc(1536 * 4);
  float*    gateOut= (float*)alloc(1536 * 4);
  float*    gateLp = (float*)alloc(1536 * 4);
  _Float16* topH   = (_Float16*)alloc((size_t)1024 * NH * 2);
  _Float16* predsH = (_Float16*)alloc((size_t)1024 * NH * 2);
  float*    predF  = (float*)alloc((size_t)1024 * NH * 4);
  _Float16* predH  = (_Float16*)alloc((size_t)1024 * NH * 2);
  float*    argF   = (float*)alloc((size_t)1024 * NH * 4);
  _Float16* argH   = (_Float16*)alloc((size_t)1024 * NH * 2);
  _Float16* UT     = (_Float16*)alloc((size_t)NL_ * NH * NH * 2);
  float*    out    = (float*)d_out;

  // 1) LN + embed concat
  k_ln_embed<<<1024, 128, 0, stream>>>(F(0), I(1), F(2), F(28), F(29), F(27), embH);

  // 2) bottom alternating LSTM (4 layers): params at 30 + 3*li
  for (int li = 0; li < 4; ++li) {
    int p = 30 + 3 * li;
    const _Float16* A = (li == 0) ? embH : hseqH;
    int K = (li == 0) ? 320 : NH, Kb = (li == 0) ? 300 : NH;
    k_gemm_f16<<<dim3(3072 / 64, 1024 / 64), 128, 0, stream>>>(
        A, F(p), nullptr, px, nullptr, 1024, 3072, K, Kb, 0);
    k_transpose_f16<<<dim3(2560 / 32, NH / 32, 1), 256, 0, stream>>>(
        F(p + 1), WsT, NH, 2560);
    k_lstm_scan<<<1, 512, 0, stream>>>(WsT, F(p + 2), px, hseqF, hseqH, li & 1);
  }

  // 3) GCN input = concat(lstm_out, constituents)
  k_build_flat<<<(1536 * NH + 255) / 256, 256, 0, stream>>>(hseqF, F(3), gF0, gH);

  // 4) three GCN layers: {param base, W_loop, W_loop_g, arc base, mask base, loop}
  const int gp[3][6] = {{48, -1, -1, 8, 20, 0}, {56, 64, 65, 12, 22, 1},
                        {66, -1, -1, 16, 25, 0}};
  float* gcur = gF0;
  float* gnxt = gF1;
  for (int gl = 0; gl < 3; ++gl) {
    int pb = gp[gl][0], ab = gp[gl][3], mb = gp[gl][4], hasl = gp[gl][5];
    dim3 gg(NH / 64, 1536 / 64);
    k_gemm_f16<<<gg, 128, 0, stream>>>(gH, F(pb + 0), nullptr, potIn, nullptr,
                                       1536, NH, NH, NH, 0);
    k_gemm_f16<<<gg, 128, 0, stream>>>(gH, F(pb + 2), nullptr, potOut, nullptr,
                                       1536, NH, NH, NH, 0);
    if (hasl)
      k_gemm_f16<<<gg, 128, 0, stream>>>(gH, F(gp[gl][1]), nullptr, potLp, nullptr,
                                         1536, NH, NH, NH, 0);
    k_matvec512<<<1536, 32, 0, stream>>>(gcur, F(pb + 4), gateIn);
    k_matvec512<<<1536, 32, 0, stream>>>(gcur, F(pb + 6), gateOut);
    if (hasl) k_matvec512<<<1536, 32, 0, stream>>>(gcur, F(gp[gl][2]), gateLp);
    k_gcn_combine<<<1536, 128, 0, stream>>>(
        gcur, potIn, potOut, hasl ? potLp : nullptr, F(pb + 1), F(pb + 3), gateIn,
        gateOut, hasl ? gateLp : nullptr, F(pb + 5), F(pb + 7), I(ab + 0), I(ab + 1),
        I(ab + 2), I(ab + 3), F(mb + 0), F(mb + 1), F(2), F(4), gnxt, gH, hasl);
    float* t = gcur; gcur = gnxt; gnxt = t;
  }

  // 5) top alternating LSTM (2 layers) on g[:, :T]
  k_gather_topin<<<(1024 * NH + 255) / 256, 256, 0, stream>>>(gH, topH);
  for (int li = 0; li < 2; ++li) {
    int p = 42 + 3 * li;
    const _Float16* A = (li == 0) ? topH : hseqH;
    k_gemm_f16<<<dim3(3072 / 64, 1024 / 64), 128, 0, stream>>>(
        A, F(p), nullptr, px, nullptr, 1024, 3072, NH, NH, 0);
    k_transpose_f16<<<dim3(2560 / 32, NH / 32, 1), 256, 0, stream>>>(
        F(p + 1), WsT, NH, 2560);
    k_lstm_scan<<<1, 512, 0, stream>>>(WsT, F(p + 2), px, hseqF, hseqH, li & 1);
  }

  // 6) predicate gather + projections (relu)
  k_gather_preds<<<(1024 * NH + 255) / 256, 256, 0, stream>>>(hseqH, I(5), predsH);
  k_gemm_f16<<<dim3(NH / 64, 1024 / 64), 128, 0, stream>>>(
      predsH, F(74), F(75), predF, predH, 1024, NH, NH, NH, 1);
  k_gemm_f16<<<dim3(NH / 64, 1024 / 64), 128, 0, stream>>>(
      hseqH, F(76), F(77), argF, argH, 1024, NH, NH, NH, 1);

  // 7) bilinear scorer + constraints
  k_bil_linear<<<1024, 64, 0, stream>>>(predF, argF, F(79), F(80), F(81), out);
  k_transpose_f16<<<dim3(NH / 32, NH / 32, NL_), 256, 0, stream>>>(F(78), UT, NH, NH);
  k_bilinear<<<dim3(1024 / 16, NL_), 128, 0, stream>>>(predH, UT, argF, I(6), out);
}
